// SearchTransfer_1434519077637
// MI455X (gfx1250) — matrix operand order repaired and checked by reading, not compiled, not run on hardware
//
#include <hip/hip_runtime.h>
#include <hip/hip_bf16.h>

typedef __attribute__((ext_vector_type(16))) _Float16 v16h;
typedef __attribute__((ext_vector_type(8)))  _Float16 v8h;
typedef __attribute__((ext_vector_type(8)))  float    v8f;

#define BATCH 2
#define C_IN  16
#define H_IN  96
#define W_IN  96
#define L_TOT (H_IN * W_IN)        // 9216 patch positions
#define K_F   144                  // C*3*3 unfolded feature length
#define KP    160                  // K padded to 5 * 32 for WMMA k-steps
#define KSTEPS 5
#define NT    (L_TOT / 16)         // 576 row (ref) tiles
#define MCHUNK (NT / 8)            // 72 row tiles per wave
#define NGRP  (L_TOT / 32)         // 288 column groups (32 cols each) per batch
#define H2    192
#define W2    192

// ---------------------------------------------------------------------------
// Phase 1: unfold(3x3, pad 1, stride 1) + L2-normalize each patch vector,
// write f16 in K-major [b][l][KP] layout (zero-padded K 144..159).
// ---------------------------------------------------------------------------
__global__ void prep_normalize(const float* __restrict__ ref_in,
                               const float* __restrict__ lr_in,
                               _Float16* __restrict__ refN,
                               _Float16* __restrict__ lrN)
{
    int gid = blockIdx.x * blockDim.x + threadIdx.x;
    if (gid >= 2 * BATCH * L_TOT) return;
    int which = gid / (BATCH * L_TOT);
    int rem   = gid % (BATCH * L_TOT);
    int b = rem / L_TOT;
    int l = rem % L_TOT;
    const float*  src = which ? lr_in : ref_in;
    _Float16*     dst = which ? lrN   : refN;

    int y = l / W_IN, x = l % W_IN;
    const float* base = src + (size_t)b * C_IN * H_IN * W_IN;

    float ss = 0.f;
    for (int c = 0; c < C_IN; ++c)
        for (int dy = 0; dy < 3; ++dy) {
            int yy = y - 1 + dy;
            for (int dx = 0; dx < 3; ++dx) {
                int xx = x - 1 + dx;
                float v = (yy >= 0 && yy < H_IN && xx >= 0 && xx < W_IN)
                            ? base[(c * H_IN + yy) * W_IN + xx] : 0.f;
                ss += v * v;
            }
        }
    float inv = 1.f / fmaxf(sqrtf(ss), 1e-12f);

    _Float16* drow = dst + ((size_t)(b * L_TOT) + l) * KP;
    int f = 0;
    for (int c = 0; c < C_IN; ++c)
        for (int dy = 0; dy < 3; ++dy) {
            int yy = y - 1 + dy;
            for (int dx = 0; dx < 3; ++dx, ++f) {
                int xx = x - 1 + dx;
                float v = (yy >= 0 && yy < H_IN && xx >= 0 && xx < W_IN)
                            ? base[(c * H_IN + yy) * W_IN + xx] : 0.f;
                drow[f] = (_Float16)(v * inv);
            }
        }
    for (; f < KP; ++f) drow[f] = (_Float16)0.f;
}

// ---------------------------------------------------------------------------
// Phase 2: fused correlation GEMM (M=N=9216, K=144) + argmax over M.
// Block = 8 waves, owns 32 lr columns. Each wave keeps two 16-column
// B-fragment sets resident (80 VGPRs) and sweeps 72 ref row tiles
// (its M-chunk); every A fragment feeds two independent WMMAs.
// Partial argmaxes merge: lane xor-16, then 8-way across waves via LDS.
// ---------------------------------------------------------------------------
__global__ void __launch_bounds__(256)
corr_argmax(const _Float16* __restrict__ refN,
            const _Float16* __restrict__ lrN,
            int* __restrict__ S,
            float* __restrict__ Sout)
{
    __shared__ float lv[8][32];
    __shared__ int   li[8][32];

    int tid    = threadIdx.x;
    int w      = tid >> 5;            // wave in block = M-chunk id
    int lane   = tid & 31;
    int laneLo = lane & 15;
    int hi     = lane >> 4;

    int b    = blockIdx.x / NGRP;
    int grp  = blockIdx.x % NGRP;
    int col0 = grp * 32;

    // Two B-fragment sets: columns [col0, col0+16) and [col0+16, col0+32).
    // Per set/k-step: lanes 0-15 hold K = s*32+[0,16), lanes 16-31 K = s*32+[16,32).
    v16h bfrag0[KSTEPS], bfrag1[KSTEPS];
    {
        const _Float16* b0 =
            lrN + ((size_t)(b * L_TOT + col0 + laneLo)) * KP + hi * 16;
        const _Float16* b1 = b0 + (size_t)16 * KP;
#pragma unroll
        for (int s = 0; s < KSTEPS; ++s) {
            const _Float16* p0 = b0 + s * 32;
            const _Float16* p1 = b1 + s * 32;
            v8h a0 = *(const v8h*)(p0);
            v8h a1 = *(const v8h*)(p0 + 8);
            v8h c0 = *(const v8h*)(p1);
            v8h c1 = *(const v8h*)(p1 + 8);
            v16h t0, t1;
#pragma unroll
            for (int e = 0; e < 8; ++e) {
                t0[e] = a0[e]; t0[8 + e] = a1[e];
                t1[e] = c0[e]; t1[8 + e] = c1[e];
            }
            bfrag0[s] = t0;
            bfrag1[s] = t1;
        }
    }

    float bV0 = -3.4e38f, bV1 = -3.4e38f;
    int   bI0 = 0,        bI1 = 0;

    const _Float16* abase0 = refN + (size_t)b * L_TOT * KP;
    int mt0 = w * MCHUNK;

    for (int mt = mt0; mt < mt0 + MCHUNK; ++mt) {
        int m0 = mt * 16;
        v8f c0 = {};
        v8f c1 = {};
        // A fragment: row (m0+laneLo); lanes 0-15 hold K [0,16),
        // lanes 16-31 hold [16,32) within each k-step.
        const _Float16* arow =
            abase0 + ((size_t)(m0 + laneLo)) * KP + hi * 16;
#pragma unroll
        for (int s = 0; s < KSTEPS; ++s) {
            const _Float16* p = arow + s * 32;
            v8h lo = *(const v8h*)(p);
            v8h hh = *(const v8h*)(p + 8);
            v16h a;
#pragma unroll
            for (int e = 0; e < 8; ++e) { a[e] = lo[e]; a[8 + e] = hh[e]; }
            c0 = __builtin_amdgcn_wmma_f32_16x16x32_f16(
                     false, a, false, bfrag0[s], (short)0, c0, false, false);
            c1 = __builtin_amdgcn_wmma_f32_16x16x32_f16(
                     false, a, false, bfrag1[s], (short)0, c1, false, false);
        }
        // C layout: VGPR r, lanes 0-15 -> M = m0+r ; lanes 16-31 -> M = m0+8+r
        int rowOff = m0 + hi * 8;
#pragma unroll
        for (int r = 0; r < 8; ++r) {
            float v0 = c0[r];
            if (v0 > bV0) { bV0 = v0; bI0 = rowOff + r; }
            float v1 = c1[r];
            if (v1 > bV1) { bV1 = v1; bI1 = rowOff + r; }
        }
    }

    // Merge the two half-wave candidates per column (lane ^ 16); scan order
    // guarantees earlier index kept within a lane, tie -> smaller index here.
    {
        float ov = __shfl_xor(bV0, 16, 32);
        int   oi = __shfl_xor(bI0, 16, 32);
        if (ov > bV0 || (ov == bV0 && oi < bI0)) { bV0 = ov; bI0 = oi; }
        ov = __shfl_xor(bV1, 16, 32);
        oi = __shfl_xor(bI1, 16, 32);
        if (ov > bV1 || (ov == bV1 && oi < bI1)) { bV1 = ov; bI1 = oi; }
    }

    if (hi == 0) {
        lv[w][laneLo]      = bV0;  li[w][laneLo]      = bI0;
        lv[w][16 + laneLo] = bV1;  li[w][16 + laneLo] = bI1;
    }
    __syncthreads();

    // 8-way merge across M-chunks; chunk order ascending in index, so
    // earlier chunk wins ties (matches argmax first-occurrence).
    if (tid < 32) {
        float bv = lv[0][tid];
        int   bi = li[0][tid];
#pragma unroll
        for (int j = 1; j < 8; ++j) {
            float v = lv[j][tid];
            int   i = li[j][tid];
            if (v > bv || (v == bv && i < bi)) { bv = v; bi = i; }
        }
        int n = b * L_TOT + col0 + tid;
        S[n]    = bi;
        Sout[n] = (float)bi;
    }
}

// ---------------------------------------------------------------------------
// Phase 3: gather 6x6 patches of org by S and fold (overlap-add, stride 2,
// pad 2) -> output-centric: each output pixel sums its <=9 covering patches.
// ---------------------------------------------------------------------------
__global__ void gather_fold(const float* __restrict__ org,
                            const int* __restrict__ S,
                            float* __restrict__ out)
{
    int gid = blockIdx.x * blockDim.x + threadIdx.x;
    if (gid >= BATCH * C_IN * H2 * W2) return;
    int x = gid % W2;
    int t = gid / W2;
    int y = t % H2; t /= H2;
    int c = t % C_IN;
    int b = t / C_IN;

    const float* ob = org + ((size_t)(b * C_IN + c)) * H2 * W2;
    const int*   Sb = S + b * L_TOT;

    float acc = 0.f;
    int lyc = (y + 2) >> 1;
    int lxc = (x + 2) >> 1;
#pragma unroll
    for (int ty = 0; ty < 3; ++ty) {
        int ly = lyc - ty;
        if ((unsigned)ly >= (unsigned)H_IN) continue;
        int dy = y + 2 - 2 * ly;          // 0..5 by construction
#pragma unroll
        for (int tx = 0; tx < 3; ++tx) {
            int lx = lxc - tx;
            if ((unsigned)lx >= (unsigned)W_IN) continue;
            int dx = x + 2 - 2 * lx;      // 0..5
            int sidx = Sb[ly * W_IN + lx];
            int sy = sidx / W_IN;
            int sx = sidx - sy * W_IN;
            int ry = 2 * sy - 2 + dy;     // source row in org
            int rx = 2 * sx - 2 + dx;
            if ((unsigned)ry < (unsigned)H2 && (unsigned)rx < (unsigned)W2)
                acc += ob[ry * W2 + rx];
        }
    }
    out[gid] = acc;
}

// ---------------------------------------------------------------------------
extern "C" void kernel_launch(void* const* d_in, const int* in_sizes, int n_in,
                              void* d_out, int out_size, void* d_ws, size_t ws_size,
                              hipStream_t stream)
{
    const float* lrsr  = (const float*)d_in[0];
    const float* refsr = (const float*)d_in[1];
    const float* org   = (const float*)d_in[2];
    // kernel_size / padding / stride / lv are compile-time constants here.

    char* ws = (char*)d_ws;
    size_t featBytes = (size_t)BATCH * L_TOT * KP * sizeof(_Float16); // 5.9 MB
    _Float16* refN = (_Float16*)ws;
    _Float16* lrN  = (_Float16*)(ws + featBytes);
    int*      S    = (int*)(ws + 2 * featBytes);

    float* out  = (float*)d_out;
    float* outS = out + (size_t)BATCH * C_IN * H2 * W2;

    // Phase 1: 36864 threads
    prep_normalize<<<(2 * BATCH * L_TOT + 255) / 256, 256, 0, stream>>>(
        refsr, lrsr, refN, lrN);

    // Phase 2: one block per (batch, 32-column group): 576 blocks, 4608 waves
    corr_argmax<<<BATCH * NGRP, 256, 0, stream>>>(refN, lrN, S, outS);

    // Phase 3: 1.18M output pixels
    gather_fold<<<(BATCH * C_IN * H2 * W2 + 255) / 256, 256, 0, stream>>>(
        org, S, out);
}
